// ConvAttBlock_26474178412816
// MI455X (gfx1250) — matrix operand order repaired and checked by reading, not compiled, not run on hardware
//
#include <hip/hip_runtime.h>

// ---------------------------------------------------------------------------
// Types / helpers
// ---------------------------------------------------------------------------
typedef __attribute__((ext_vector_type(16))) __bf16         bf16x16;
typedef __attribute__((ext_vector_type(8)))  float          f32x8;
typedef __attribute__((ext_vector_type(4)))  unsigned int   u32x4;
typedef __attribute__((ext_vector_type(8)))  int            i32x8;
typedef __attribute__((ext_vector_type(4)))  int            i32x4;

__device__ __forceinline__ __bf16 f2bf(float f) {
    unsigned u = __builtin_bit_cast(unsigned, f);
    u += 0x7fffu + ((u >> 16) & 1u);          // round-to-nearest-even
    unsigned short h = (unsigned short)(u >> 16);
    return __builtin_bit_cast(__bf16, h);
}
__device__ __forceinline__ float bf2f(__bf16 h) {
    unsigned short us = __builtin_bit_cast(unsigned short, h);
    unsigned u = ((unsigned)us) << 16;
    return __builtin_bit_cast(float, u);
}

// A-fragment K offset for v_wmma_*_16x16x32_bf16 (contiguous k = 16*hi + e, matches B fills)
__device__ __forceinline__ int akoff(int e, int hi) {
    return e + 16 * hi;
}

__device__ __forceinline__ f32x8 wmma_bf16(bf16x16 a, bf16x16 b, f32x8 c) {
    return __builtin_amdgcn_wmma_f32_16x16x32_bf16(
        false, a, false, b, (short)0, c, false, false);
}

// Raw LDS byte offset of a __shared__ object (addrspace(3) pointers are the
// 32-bit LDS offset on amdgcn).
__device__ __forceinline__ unsigned ldsAddr(const void* p) {
    return (unsigned)(size_t)(__attribute__((address_space(3))) const char*)p;
}

// Tensor Data Mover: async 2-D tile load (bf16) global -> LDS.
// Tile: tileRows x tileCols elements, row stride rowStrideElems in global;
// packed contiguously in LDS at lds_off. Descriptor per CDNA5 ISA 8.3/8.4.
__device__ __forceinline__ void tdm_load_tile_2d(
    const void* gsrc, unsigned lds_off,
    unsigned tileRows, unsigned tileCols, unsigned rowStrideElems)
{
    unsigned long long ga = (unsigned long long)(size_t)gsrc;
    u32x4 g0;
    g0[0] = 1u;                                   // count=1 (valid), user mode
    g0[1] = lds_off;                              // lds_addr (bytes)
    g0[2] = (unsigned)ga;                         // global_addr[31:0]
    g0[3] = (unsigned)((ga >> 32) & 0x1ffffffull) // global_addr[56:32]
          | (2u << 30);                           // type=2 ("image")
    i32x8 g1;
    g1[0] = (int)(1u << 16);                      // data_size=1 -> 2 bytes
    g1[1] = (int)(tileCols << 16);                // tensor_dim0[15:0]
    g1[2] = (int)(tileRows << 16);                // tensor_dim1[15:0]
    g1[3] = (int)(tileCols << 16);                // tile_dim0
    g1[4] = (int)tileRows;                        // tile_dim1 (tile_dim2=0)
    g1[5] = (int)rowStrideElems;                  // tensor_dim0_stride[31:0]
    g1[6] = 0;
    g1[7] = 0;
    i32x4 gz = {0, 0, 0, 0};
#if defined(__clang_major__) && (__clang_major__ >= 23)
    i32x8 gz8 = {0, 0, 0, 0, 0, 0, 0, 0};
    __builtin_amdgcn_tensor_load_to_lds(g0, g1, gz, gz, gz8, 0);
#else
    __builtin_amdgcn_tensor_load_to_lds(g0, g1, gz, gz, 0);
#endif
}

// Problem constants
#define BB     4
#define HW     48
#define NPIX   (HW * HW)        // 2304
#define CIN    256
#define HID    64
#define CTX    512
#define MCTX   1024
#define NHEAD  8
#define DHEAD  64
#define INNER  (NHEAD * DHEAD)  // 512
#define COUT   256
#define K1     (CIN * 9)        // 2304
#define K2     (HID * 9)        // 576
#define SCALE  0.125f

// ---------------------------------------------------------------------------
// 1) Spectral norm power-iteration + pack scaled weight bf16 [O][K] (K-major)
// ---------------------------------------------------------------------------
__global__ __launch_bounds__(256)
void spectral_pack(const float* __restrict__ w, const float* __restrict__ u,
                   int O, int K, float* __restrict__ inv_sigma_out,
                   __bf16* __restrict__ Bout)
{
    __shared__ float sv[K1];
    __shared__ float red[256];
    __shared__ float s_inv;
    const int t = threadIdx.x;

    for (int j = t; j < K; j += 256) {            // v = wm^T @ u
        float s = 0.f;
        for (int o = 0; o < O; ++o) s += w[(size_t)o * K + j] * u[o];
        sv[j] = s;
    }
    __syncthreads();

    float p = 0.f;
    for (int j = t; j < K; j += 256) p += sv[j] * sv[j];
    red[t] = p; __syncthreads();
    for (int s = 128; s > 0; s >>= 1) { if (t < s) red[t] += red[t + s]; __syncthreads(); }
    const float vinv = 1.f / (sqrtf(red[0]) + 1e-12f);
    __syncthreads();

    float q2 = 0.f;
    for (int o = t; o < O; o += 256) {            // t = wm @ v_hat
        float s = 0.f;
        for (int j = 0; j < K; ++j) s += w[(size_t)o * K + j] * sv[j];
        s *= vinv;
        q2 += s * s;
    }
    red[t] = q2; __syncthreads();
    for (int s = 128; s > 0; s >>= 1) { if (t < s) red[t] += red[t + s]; __syncthreads(); }
    if (t == 0) {
        float nt = sqrtf(red[0]);
        float sigma = (nt * nt) / (nt + 1e-12f);
        s_inv = 1.f / sigma;
        inv_sigma_out[0] = s_inv;
    }
    __syncthreads();
    const float inv = s_inv;
    for (int idx = t; idx < O * K; idx += 256)
        Bout[idx] = f2bf(w[idx] * inv);
}

// 2) flat f32 -> bf16 (weights kept in native [N][K], context flat)
__global__ __launch_bounds__(256)
void cvt_bf16(const float* __restrict__ in, __bf16* __restrict__ out, int n)
{
    int idx = blockIdx.x * 256 + threadIdx.x;
    if (idx < n) out[idx] = f2bf(in[idx]);
}

// 3) V -> V^T per batch:  in [B][MCTX][INNER] -> out [B][INNER][MCTX]
__global__ __launch_bounds__(256)
void transpose_v(const __bf16* __restrict__ in, __bf16* __restrict__ outT)
{
    int idx = blockIdx.x * 256 + threadIdx.x;
    if (idx < BB * MCTX * INNER) {
        int b = idx / (MCTX * INNER);
        int r = (idx / INNER) % MCTX;
        int d = idx % INNER;
        outT[((size_t)b * INNER + d) * MCTX + r] = in[idx];
    }
}

// ---------------------------------------------------------------------------
// 4) im2col kernels (A matrices, bf16 row-major [rows][K])
// ---------------------------------------------------------------------------
__global__ __launch_bounds__(256)
void im2col_x(const float* __restrict__ x, __bf16* __restrict__ A)
{
    const int g = blockIdx.x;
    const int b = g / NPIX, pp = g % NPIX;
    const int hh = pp / HW, ww = pp % HW;
    const float* xb = x + (size_t)b * CIN * NPIX;
    __bf16* Ar = A + (size_t)g * K1;
    for (int j = threadIdx.x; j < K1; j += 256) {
        int c = j / 9, r9 = j % 9, kh = r9 / 3, kw = r9 % 3;
        int ih = hh + kh - 1, iw = ww + kw - 1;
        float v = 0.f;
        if ((unsigned)ih < (unsigned)HW && (unsigned)iw < (unsigned)HW)
            v = xb[(size_t)c * NPIX + ih * HW + iw];
        Ar[j] = f2bf(v);
    }
}

__global__ __launch_bounds__(256)
void im2col_t(const float* __restrict__ t2, __bf16* __restrict__ A)
{
    const int g = blockIdx.x;
    const int b = g / NPIX, pp = g % NPIX;
    const int hh = pp / HW, ww = pp % HW;
    __bf16* Ar = A + (size_t)g * K2;
    for (int j = threadIdx.x; j < K2; j += 256) {
        int c = j / 9, r9 = j % 9, kh = r9 / 3, kw = r9 % 3;
        int ih = hh + kh - 1, iw = ww + kw - 1;
        float v = 0.f;
        if ((unsigned)ih < (unsigned)HW && (unsigned)iw < (unsigned)HW)
            v = t2[((size_t)b * NPIX + ih * HW + iw) * HID + c];
        Ar[j] = f2bf(v);
    }
}

// ---------------------------------------------------------------------------
// 5) WMMA GEMM: D[M][N] = A[M][K] * Bt[N][K]^T   (both bf16, K-contiguous)
//    128 threads = 4 waves; block tile 64x64; K-step 32.
//    TDM double-buffered LDS staging (wave 0 drives the DMA).
// ---------------------------------------------------------------------------
__global__ __launch_bounds__(128)
void gemm_bf16_wmma(const __bf16* __restrict__ A, const __bf16* __restrict__ Bt,
                    float* __restrict__ Df, __bf16* __restrict__ Db,
                    const float* __restrict__ bias, int M, int N, int K)
{
    __shared__ __align__(16) __bf16 As[2][64 * 32];
    __shared__ __align__(16) __bf16 Bs[2][64 * 32];
    const int tid  = threadIdx.x;
    const int lane = tid & 31, wid = tid >> 5;
    const int ml   = lane & 15, hi = lane >> 4;
    const int rowBase = blockIdx.y * 64;
    const int colBase = blockIdx.x * 64;

    f32x8 acc[4];
    acc[0] = f32x8{}; acc[1] = f32x8{}; acc[2] = f32x8{}; acc[3] = f32x8{};

    const int steps = K / 32;
    const __bf16* Abase = A  + (size_t)rowBase * K;
    const __bf16* Bbase = Bt + (size_t)colBase * K;

    if (tid < 32) {
        tdm_load_tile_2d(Abase, ldsAddr(&As[0][0]), 64, 32, (unsigned)K);
        tdm_load_tile_2d(Bbase, ldsAddr(&Bs[0][0]), 64, 32, (unsigned)K);
    }

    for (int s = 0; s < steps; ++s) {
        const int cur = s & 1;
        if (tid < 32) {
            if (s + 1 < steps) {
                tdm_load_tile_2d(Abase + (s + 1) * 32, ldsAddr(&As[cur ^ 1][0]),
                                 64, 32, (unsigned)K);
                tdm_load_tile_2d(Bbase + (s + 1) * 32, ldsAddr(&Bs[cur ^ 1][0]),
                                 64, 32, (unsigned)K);
                __builtin_amdgcn_s_wait_tensorcnt((short)2);
            } else {
                __builtin_amdgcn_s_wait_tensorcnt((short)0);
            }
        }
        __syncthreads();

        bf16x16 af;
#pragma unroll
        for (int e = 0; e < 16; ++e)
            af[e] = As[cur][(wid * 16 + ml) * 32 + akoff(e, hi)];

#pragma unroll
        for (int cs = 0; cs < 4; ++cs) {
            bf16x16 bfr;
#pragma unroll
            for (int e = 0; e < 16; ++e)
                bfr[e] = Bs[cur][(cs * 16 + ml) * 32 + 16 * hi + e];
            acc[cs] = wmma_bf16(af, bfr, acc[cs]);
        }
        __syncthreads();
    }

#pragma unroll
    for (int cs = 0; cs < 4; ++cs) {
        const int col = colBase + cs * 16 + ml;
        const float bv = bias ? bias[col] : 0.f;
#pragma unroll
        for (int r = 0; r < 8; ++r) {
            const int row = rowBase + wid * 16 + 8 * hi + r;
            const float v = acc[cs][r] + bv;
            if (Df) Df[(size_t)row * N + col] = v;
            else    Db[(size_t)row * N + col] = f2bf(v);
        }
    }
}

// ---------------------------------------------------------------------------
// 6) Cross-attention, one wave per (b, head, 16-query-row tile).
//    QK^T WMMA -> 16x1024 bf16 score sheet in LDS (32 KB) -> softmax -> P.V.
//    V consumed via V^T so PV B-fragments are contiguous global loads.
// ---------------------------------------------------------------------------
__global__ __launch_bounds__(32)
void attn_wmma(const __bf16* __restrict__ q, const __bf16* __restrict__ k,
               const __bf16* __restrict__ vt, __bf16* __restrict__ o)
{
    __shared__ __align__(16) __bf16 scb[16 * MCTX];   // 32 KB
    const int lane = threadIdx.x;
    const int ml = lane & 15, hi = lane >> 4;
    const int bh = blockIdx.y;
    const int b = bh >> 3, h = bh & 7;
    const int row0 = blockIdx.x * 16;

    const __bf16* qp = q  + ((size_t)(b * NPIX)) * INNER + h * DHEAD;
    const __bf16* kp = k  + ((size_t)(b * MCTX)) * INNER + h * DHEAD;
    const __bf16* vp = vt + ((size_t)(b * INNER) + h * DHEAD) * MCTX;

    // Q fragments for this 16-row tile (d = 0..31 and 32..63)
    bf16x16 qa0, qa1;
#pragma unroll
    for (int e = 0; e < 16; ++e) {
        const int kd = akoff(e, hi);
        qa0[e] = qp[(size_t)(row0 + ml) * INNER + kd];
        qa1[e] = qp[(size_t)(row0 + ml) * INNER + 32 + kd];
    }

    // S = (Q K^T) * SCALE  -> LDS (bf16)
    for (int kt = 0; kt < MCTX / 16; ++kt) {
        f32x8 s = f32x8{};
        bf16x16 kf;
#pragma unroll
        for (int e = 0; e < 16; ++e)
            kf[e] = kp[(size_t)(kt * 16 + ml) * INNER + (e + 16 * hi)];
        s = wmma_bf16(qa0, kf, s);
#pragma unroll
        for (int e = 0; e < 16; ++e)
            kf[e] = kp[(size_t)(kt * 16 + ml) * INNER + 32 + (e + 16 * hi)];
        s = wmma_bf16(qa1, kf, s);
#pragma unroll
        for (int r = 0; r < 8; ++r)
            scb[(r + 8 * hi) * MCTX + kt * 16 + ml] = f2bf(s[r] * SCALE);
    }
    __syncthreads();

    // softmax along keys for each of the 16 rows (whole wave per row)
    for (int r = 0; r < 16; ++r) {
        float m = -1e30f;
        for (int i = lane; i < MCTX; i += 32)
            m = fmaxf(m, bf2f(scb[r * MCTX + i]));
        for (int off = 16; off; off >>= 1) m = fmaxf(m, __shfl_xor(m, off, 32));
        float ssum = 0.f;
        for (int i = lane; i < MCTX; i += 32) {
            float e = __expf(bf2f(scb[r * MCTX + i]) - m);
            scb[r * MCTX + i] = f2bf(e);
            ssum += e;
        }
        for (int off = 16; off; off >>= 1) ssum += __shfl_xor(ssum, off, 32);
        const float inv = 1.f / ssum;
        for (int i = lane; i < MCTX; i += 32)
            scb[r * MCTX + i] = f2bf(bf2f(scb[r * MCTX + i]) * inv);
    }
    __syncthreads();

    // O = P V
    f32x8 oa[4];
    oa[0] = f32x8{}; oa[1] = f32x8{}; oa[2] = f32x8{}; oa[3] = f32x8{};
    for (int kkk = 0; kkk < MCTX; kkk += 32) {
        bf16x16 pf;
#pragma unroll
        for (int e = 0; e < 16; ++e)
            pf[e] = scb[ml * MCTX + kkk + akoff(e, hi)];
#pragma unroll
        for (int cs = 0; cs < 4; ++cs) {
            bf16x16 vf;
#pragma unroll
            for (int e = 0; e < 16; ++e)
                vf[e] = vp[(size_t)(cs * 16 + ml) * MCTX + kkk + 16 * hi + e];
            oa[cs] = wmma_bf16(pf, vf, oa[cs]);
        }
    }

    __bf16* op = o + ((size_t)(b * NPIX + row0)) * INNER + h * DHEAD;
#pragma unroll
    for (int cs = 0; cs < 4; ++cs)
#pragma unroll
        for (int r = 0; r < 8; ++r)
            op[(size_t)(r + 8 * hi) * INNER + cs * 16 + ml] = f2bf(oa[cs][r]);
}

// ---------------------------------------------------------------------------
// 7) Instance norm + ReLU. One block per (b, c) plane (2304 pixels).
// ---------------------------------------------------------------------------
__global__ __launch_bounds__(256)
void inorm_relu_tokens(const float* __restrict__ in, __bf16* __restrict__ outb, int C)
{
    const int b = blockIdx.x / C, c = blockIdx.x % C;
    const float* base = in + (size_t)b * NPIX * C + c;
    __bf16* ob = outb + (size_t)b * NPIX * C + c;
    __shared__ float r1[256], r2[256];
    const int t = threadIdx.x;
    float s = 0.f, s2 = 0.f;
    for (int p = t; p < NPIX; p += 256) {
        float x = base[(size_t)p * C];
        s += x; s2 += x * x;
    }
    r1[t] = s; r2[t] = s2; __syncthreads();
    for (int st = 128; st > 0; st >>= 1) {
        if (t < st) { r1[t] += r1[t + st]; r2[t] += r2[t + st]; }
        __syncthreads();
    }
    const float mu  = r1[0] / (float)NPIX;
    const float var = r2[0] / (float)NPIX - mu * mu;
    const float rs  = rsqrtf(var + 1e-5f);
    for (int p = t; p < NPIX; p += 256) {
        float y = (base[(size_t)p * C] - mu) * rs;
        ob[(size_t)p * C] = f2bf(fmaxf(y, 0.f));
    }
}

__global__ __launch_bounds__(256)
void inorm_relu_nchw(const float* __restrict__ in, float* __restrict__ out, int C)
{
    const int b = blockIdx.x / C, c = blockIdx.x % C;
    const float* base = in + (size_t)b * NPIX * C + c;
    float* ob = out + ((size_t)b * C + c) * NPIX;
    __shared__ float r1[256], r2[256];
    const int t = threadIdx.x;
    float s = 0.f, s2 = 0.f;
    for (int p = t; p < NPIX; p += 256) {
        float x = base[(size_t)p * C];
        s += x; s2 += x * x;
    }
    r1[t] = s; r2[t] = s2; __syncthreads();
    for (int st = 128; st > 0; st >>= 1) {
        if (t < st) { r1[t] += r1[t + st]; r2[t] += r2[t + st]; }
        __syncthreads();
    }
    const float mu  = r1[0] / (float)NPIX;
    const float var = r2[0] / (float)NPIX - mu * mu;
    const float rs  = rsqrtf(var + 1e-5f);
    for (int p = t; p < NPIX; p += 256) {
        float y = (base[(size_t)p * C] - mu) * rs;
        ob[p] = fmaxf(y, 0.f);
    }
}

// ---------------------------------------------------------------------------
// Host orchestration
// ---------------------------------------------------------------------------
extern "C" void kernel_launch(void* const* d_in, const int* in_sizes, int n_in,
                              void* d_out, int out_size, void* d_ws, size_t ws_size,
                              hipStream_t stream)
{
    const float* x   = (const float*)d_in[0];
    const float* ctx = (const float*)d_in[1];
    const float* w1  = (const float*)d_in[2];
    const float* u1  = (const float*)d_in[3];
    const float* wq  = (const float*)d_in[4];
    const float* wk  = (const float*)d_in[5];
    const float* wv  = (const float*)d_in[6];
    const float* wo  = (const float*)d_in[7];
    const float* bo  = (const float*)d_in[8];
    const float* w2  = (const float*)d_in[9];
    const float* u2  = (const float*)d_in[10];
    float* out = (float*)d_out;

    char* p = (char*)d_ws;
    auto carve = [&](size_t bytes) -> char* {
        char* r = p;
        p += (bytes + 255) & ~(size_t)255;
        return r;
    };
    const int NTOK = BB * NPIX;   // 9216
    const int NCTX = BB * MCTX;   // 4096

    float*  invs   = (float*) carve(256);
    __bf16* w1s    = (__bf16*)carve((size_t)HID  * K1 * 2);   // [N][K]
    __bf16* w2s    = (__bf16*)carve((size_t)COUT * K2 * 2);
    __bf16* wqb    = (__bf16*)carve((size_t)INNER * HID * 2);
    __bf16* wkb    = (__bf16*)carve((size_t)INNER * CTX * 2);
    __bf16* wvb    = (__bf16*)carve((size_t)INNER * CTX * 2);
    __bf16* wob    = (__bf16*)carve((size_t)HID * INNER * 2);
    __bf16* ctxb   = (__bf16*)carve((size_t)NCTX * CTX * 2);
    __bf16* A1     = (__bf16*)carve((size_t)NTOK * K1 * 2);
    float*  c1raw  = (float*) carve((size_t)NTOK * HID * 4);
    __bf16* tok    = (__bf16*)carve((size_t)NTOK * HID * 2);
    __bf16* qb     = (__bf16*)carve((size_t)NTOK * INNER * 2);
    __bf16* kb     = (__bf16*)carve((size_t)NCTX * INNER * 2);
    __bf16* vb     = (__bf16*)carve((size_t)NCTX * INNER * 2);
    __bf16* vtb    = (__bf16*)carve((size_t)NCTX * INNER * 2);
    __bf16* attnb  = (__bf16*)carve((size_t)NTOK * INNER * 2);
    float*  tok2   = (float*) carve((size_t)NTOK * HID * 4);
    __bf16* A2     = (__bf16*)carve((size_t)NTOK * K2 * 2);
    float*  c2raw  = (float*) carve((size_t)NTOK * COUT * 4);

    // 1) spectral-norm + pack conv weights ([O][K], K-contiguous)
    spectral_pack<<<1, 256, 0, stream>>>(w1, u1, HID,  K1, invs + 0, w1s);
    spectral_pack<<<1, 256, 0, stream>>>(w2, u2, COUT, K2, invs + 1, w2s);

    // 2) projection weights + context -> bf16 (native [N][K] layout kept)
    cvt_bf16<<<(INNER * HID + 255) / 256, 256, 0, stream>>>(wq, wqb, INNER * HID);
    cvt_bf16<<<(INNER * CTX + 255) / 256, 256, 0, stream>>>(wk, wkb, INNER * CTX);
    cvt_bf16<<<(INNER * CTX + 255) / 256, 256, 0, stream>>>(wv, wvb, INNER * CTX);
    cvt_bf16<<<(HID * INNER + 255) / 256, 256, 0, stream>>>(wo, wob, HID * INNER);
    cvt_bf16<<<(NCTX * CTX + 255) / 256, 256, 0, stream>>>(ctx, ctxb, NCTX * CTX);

    // 3) conv1 as implicit GEMM
    im2col_x<<<NTOK, 256, 0, stream>>>(x, A1);
    gemm_bf16_wmma<<<dim3(HID / 64, NTOK / 64), 128, 0, stream>>>(
        A1, w1s, c1raw, nullptr, nullptr, NTOK, HID, K1);
    inorm_relu_tokens<<<BB * HID, 256, 0, stream>>>(c1raw, tok, HID);

    // 4) Q/K/V projections
    gemm_bf16_wmma<<<dim3(INNER / 64, NTOK / 64), 128, 0, stream>>>(
        tok, wqb, nullptr, qb, nullptr, NTOK, INNER, HID);
    gemm_bf16_wmma<<<dim3(INNER / 64, NCTX / 64), 128, 0, stream>>>(
        ctxb, wkb, nullptr, kb, nullptr, NCTX, INNER, CTX);
    gemm_bf16_wmma<<<dim3(INNER / 64, NCTX / 64), 128, 0, stream>>>(
        ctxb, wvb, nullptr, vb, nullptr, NCTX, INNER, CTX);
    transpose_v<<<(NCTX * INNER + 255) / 256, 256, 0, stream>>>(vb, vtb);

    // 5) attention
    attn_wmma<<<dim3(NPIX / 16, BB * NHEAD), 32, 0, stream>>>(qb, kb, vtb, attnb);

    // 6) output projection (+bias)
    gemm_bf16_wmma<<<dim3(HID / 64, NTOK / 64), 128, 0, stream>>>(
        attnb, wob, tok2, nullptr, bo, NTOK, HID, INNER);

    // 7) conv2 as implicit GEMM + IN + ReLU -> d_out (NCHW f32)
    im2col_t<<<NTOK, 256, 0, stream>>>(tok2, A2);
    gemm_bf16_wmma<<<dim3(COUT / 64, NTOK / 64), 128, 0, stream>>>(
        A2, w2s, c2raw, nullptr, nullptr, NTOK, COUT, K2);
    inorm_relu_nchw<<<BB * COUT, 256, 0, stream>>>(c2raw, out, COUT);
}
